// FusionModule_69715909149185
// MI455X (gfx1250) — matrix operand order repaired and checked by reading, not compiled, not run on hardware
//
#include <hip/hip_runtime.h>
#include <hip/hip_bf16.h>

typedef _Float16 f16;
typedef _Float16 v16h __attribute__((ext_vector_type(16)));
typedef _Float16 v8h  __attribute__((ext_vector_type(8)));
typedef float    v8f  __attribute__((ext_vector_type(8)));

// ---------------------------------------------------------------- elementwise
__global__ void add_kernel(const float* __restrict__ a, const float* __restrict__ b,
                           float* __restrict__ out, int n) {
  int i = blockIdx.x * blockDim.x + threadIdx.x;
  if (i < n) out[i] = a[i] + b[i];
}

// ------------------------------------------------------- direct offset conv
__global__ void conv_offset_kernel(const float* __restrict__ x, const float* __restrict__ w,
                                   const float* __restrict__ bias, float* __restrict__ out,
                                   int B, int Cin, int H, int W, int Coff, int Ho, int Wo,
                                   int k, int stride, int pad) {
  int idx = blockIdx.x * blockDim.x + threadIdx.x;
  int total = B * Coff * Ho * Wo;
  if (idx >= total) return;
  int wo = idx % Wo;
  int ho = (idx / Wo) % Ho;
  int oc = (idx / (Wo * Ho)) % Coff;
  int bb = idx / (Wo * Ho * Coff);
  float acc = bias[oc];
  for (int ci = 0; ci < Cin; ++ci) {
    const float* xp = x + ((size_t)(bb * Cin + ci) * H) * W;
    const float* wp = w + ((size_t)(oc * Cin + ci) * k) * k;
    for (int ky = 0; ky < k; ++ky) {
      int y = ho * stride - pad + ky;
      if (y < 0 || y >= H) continue;
      for (int kx = 0; kx < k; ++kx) {
        int xx = wo * stride - pad + kx;
        if (xx < 0 || xx >= W) continue;
        acc += xp[y * W + xx] * wp[ky * k + kx];
      }
    }
  }
  out[((size_t)(bb * Coff + oc) * Ho + ho) * Wo + wo] = acc;
}

// ------------------------------------- deformable bilinear gather -> im2col f16
// V layout: [n = b*Ho*Wo + hw][ci*KK + kk]  (n-major, K contiguous per row)
__global__ void dcn_gather_kernel(const float* __restrict__ x, const float* __restrict__ om,
                                  f16* __restrict__ V,
                                  int B, int Cin, int H, int W, int Ho, int Wo,
                                  int k, int stride, int pad) {
  int KK = k * k;
  int HoWo = Ho * Wo;
  int total = B * HoWo * KK;
  int idx = blockIdx.x * blockDim.x + threadIdx.x;
  if (idx >= total) return;
  int kk = idx % KK;
  int n  = idx / KK;
  int hw = n % HoWo;
  int bb = n / HoWo;
  int ho = hw / Wo, wo = hw % Wo;

  size_t ombase = (size_t)bb * 3 * KK * HoWo + hw;
  float oy = om[ombase + (size_t)kk * HoWo];
  float ox = om[ombase + (size_t)(KK + kk) * HoWo];
  float mm = om[ombase + (size_t)(2 * KK + kk) * HoWo];
  mm = 1.0f / (1.0f + expf(-mm));

  int ky = kk / k, kx = kk % k;
  float sy = (float)(ky + ho * stride - pad) + oy;
  float sx = (float)(kx + wo * stride - pad) + ox;
  float y0f = floorf(sy), x0f = floorf(sx);
  float ly = sy - y0f, lx = sx - x0f;
  int y0 = (int)y0f, x0 = (int)x0f;
  int y1 = y0 + 1, x1 = x0 + 1;
  float vy0 = (y0 >= 0 && y0 < H) ? 1.f : 0.f;
  float vy1 = (y1 >= 0 && y1 < H) ? 1.f : 0.f;
  float vx0 = (x0 >= 0 && x0 < W) ? 1.f : 0.f;
  float vx1 = (x1 >= 0 && x1 < W) ? 1.f : 0.f;
  int cy0 = min(max(y0, 0), H - 1), cy1 = min(max(y1, 0), H - 1);
  int cx0 = min(max(x0, 0), W - 1), cx1 = min(max(x1, 0), W - 1);
  float w00 = (1.f - ly) * (1.f - lx) * vy0 * vx0 * mm;
  float w01 = (1.f - ly) * lx        * vy0 * vx1 * mm;
  float w10 = ly        * (1.f - lx) * vy1 * vx0 * mm;
  float w11 = ly        * lx         * vy1 * vx1 * mm;
  int i00 = cy0 * W + cx0, i01 = cy0 * W + cx1;
  int i10 = cy1 * W + cx0, i11 = cy1 * W + cx1;

  int Kd = Cin * KK;
  f16* vp = V + (size_t)n * Kd + kk;
  const float* xb = x + (size_t)bb * Cin * H * W;
  for (int ci = 0; ci < Cin; ++ci) {
    const float* xp = xb + (size_t)ci * H * W;
    float v = w00 * xp[i00] + w01 * xp[i01] + w10 * xp[i10] + w11 * xp[i11];
    vp[(size_t)ci * KK] = (f16)v;
  }
}

// ------------------------------------------------------------ f32 -> f16 weights
__global__ void convert_w_kernel(const float* __restrict__ w, f16* __restrict__ wh, int n) {
  int i = blockIdx.x * blockDim.x + threadIdx.x;
  if (i < n) wh[i] = (f16)w[i];
}

// ------------------------------------------------------------ bilinear upsample 2x
__global__ void upsample2x_kernel(const float* __restrict__ x, float* __restrict__ out,
                                  int B, int C, int H, int W) {
  int Ho = 2 * H, Wo = 2 * W;
  int total = B * C * Ho * Wo;
  int idx = blockIdx.x * blockDim.x + threadIdx.x;
  if (idx >= total) return;
  int ox = idx % Wo;
  int oy = (idx / Wo) % Ho;
  int c  = (idx / (Wo * Ho)) % C;
  int bb = idx / (Wo * Ho * C);
  float sy = (oy + 0.5f) * 0.5f - 0.5f;
  float sx = (ox + 0.5f) * 0.5f - 0.5f;
  float y0f = floorf(sy), x0f = floorf(sx);
  float ly = sy - y0f, lx = sx - x0f;
  int y0 = (int)y0f, x0 = (int)x0f;
  int y1 = y0 + 1, x1 = x0 + 1;
  y0 = min(max(y0, 0), H - 1); y1 = min(max(y1, 0), H - 1);
  x0 = min(max(x0, 0), W - 1); x1 = min(max(x1, 0), W - 1);
  const float* xp = x + ((size_t)(bb * C + c) * H) * W;
  float v = (1.f - ly) * ((1.f - lx) * xp[y0 * W + x0] + lx * xp[y0 * W + x1]) +
            ly         * ((1.f - lx) * xp[y1 * W + x0] + lx * xp[y1 * W + x1]);
  out[idx] = v;
}

// ------------------------------------------------------------ WMMA GEMM + epilogue
// Out[((b*M)+m)*HoWo + hw] = epilogue( sum_k A[m][k] * Bv[n][k] ), n = b*HoWo+hw
// mode 0: bias + BN + relu (down path).  mode 1: relu(bias+acc), then relu(res + .)
// Async double-buffered global->LDS staging (GLOBAL_LOAD_ASYNC_TO_LDS_B128, ASYNCcnt),
// steady-state loop is branch-free (final slab peeled).
#define BM 128
#define BN 128
#define BKK 32

__device__ __forceinline__ void async_b128(unsigned lds_off, const void* gaddr) {
  unsigned long long ga = (unsigned long long)gaddr;
  asm volatile("global_load_async_to_lds_b128 %0, %1, off"
               :: "v"(lds_off), "v"(ga) : "memory");
}

__global__ __launch_bounds__(256) void wmma_gemm_kernel(
    const f16* __restrict__ A, const f16* __restrict__ Bv, float* __restrict__ Out,
    const float* __restrict__ bias,
    const float* __restrict__ gamma, const float* __restrict__ beta,
    const float* __restrict__ mean, const float* __restrict__ var,
    const float* __restrict__ residual,
    int M, int N, int Kd, int HoWo, int mode) {
  __shared__ __align__(16) f16 sA[2][BM * BKK];
  __shared__ __align__(16) f16 sB[2][BN * BKK];

  int t = threadIdx.x;
  int lane = t & 31;
  int wave = t >> 5;
  int wm = wave >> 2;   // 0..1 : 64-row M slab
  int wn = wave & 3;    // 0..3 : 32-col N slab
  int lrow = lane & 15;
  int lhi  = lane >> 4;
  int m0 = blockIdx.y * BM;
  int n0 = blockIdx.x * BN;

  // staging chunks: 512 16B chunks per tile, 2 per thread; chunk id -> (row, col)
  int rA0 = t >> 2,        cA = (t & 3) * 8;
  int rA1 = 64 + (t >> 2);
  // clamp A rows so we always issue exactly 4 async loads per thread per step
  int mA0 = min(m0 + rA0, M - 1);
  int mA1 = min(m0 + rA1, M - 1);
  const f16* gA0 = A + (size_t)mA0 * Kd + cA;
  const f16* gA1 = A + (size_t)mA1 * Kd + cA;
  const f16* gB0 = Bv + (size_t)(n0 + rA0) * Kd + cA;
  const f16* gB1 = Bv + (size_t)(n0 + rA1) * Kd + cA;

  // low 32 bits of generic shared address == LDS byte offset
  unsigned lA0 = (unsigned)(unsigned long long)&sA[0][rA0 * BKK + cA];
  unsigned lA1 = (unsigned)(unsigned long long)&sA[0][rA1 * BKK + cA];
  unsigned lB0 = (unsigned)(unsigned long long)&sB[0][rA0 * BKK + cA];
  unsigned lB1 = (unsigned)(unsigned long long)&sB[0][rA1 * BKK + cA];
  const unsigned bufA = (unsigned)(BM * BKK * 2);  // bytes per buffer
  const unsigned bufB = (unsigned)(BN * BKK * 2);

  v8f zero = {};
  v8f acc[4][2];
#pragma unroll
  for (int mi = 0; mi < 4; ++mi)
#pragma unroll
    for (int ni = 0; ni < 2; ++ni) acc[mi][ni] = zero;

  // one K-slab of WMMA work from LDS buffer p
  auto compute_slab = [&](int p) {
    v16h afrag[4];
#pragma unroll
    for (int mi = 0; mi < 4; ++mi) {
      int arow = wm * 64 + mi * 16 + lrow;
      v8h lo = *(const v8h*)(&sA[p][arow * BKK + lhi * 8]);
      v8h hi = *(const v8h*)(&sA[p][arow * BKK + 16 + lhi * 8]);
      v16h af;
#pragma unroll
      for (int e = 0; e < 8; ++e) { af[e] = lo[e]; af[e + 8] = hi[e]; }
      afrag[mi] = af;
    }
    v16h bfrag[2];
#pragma unroll
    for (int ni = 0; ni < 2; ++ni) {
      int brow = wn * 32 + ni * 16 + lrow;
      v8h lo = *(const v8h*)(&sB[p][brow * BKK + lhi * 8]);
      v8h hi = *(const v8h*)(&sB[p][brow * BKK + 16 + lhi * 8]);
      v16h bf;
#pragma unroll
      for (int e = 0; e < 8; ++e) { bf[e] = lo[e]; bf[e + 8] = hi[e]; }
      bfrag[ni] = bf;
    }
#pragma unroll
    for (int mi = 0; mi < 4; ++mi)
#pragma unroll
      for (int ni = 0; ni < 2; ++ni)
        acc[mi][ni] = __builtin_amdgcn_wmma_f32_16x16x32_f16(
            false, afrag[mi], false, bfrag[ni], (short)0, acc[mi][ni], false, false);
  };

  // prologue: stream first K-slab into buffer 0 (4 async b128 per thread)
  async_b128(lA0, gA0);
  async_b128(lA1, gA1);
  async_b128(lB0, gB0);
  async_b128(lB1, gB1);

  unsigned oA = bufA, oB = bufB;  // offsets of the "other" buffer
  int p = 0;
  // steady state: branch-free issue -> wait(4) -> barrier -> wmma -> barrier
  for (int k0 = 0; k0 + BKK < Kd; k0 += BKK) {
    int kn = k0 + BKK;
    async_b128(lA0 + oA, gA0 + kn);
    async_b128(lA1 + oA, gA1 + kn);
    async_b128(lB0 + oB, gB0 + kn);
    async_b128(lB1 + oB, gB1 + kn);
    asm volatile("s_wait_asynccnt 0x4" ::: "memory");  // slab for buf p complete
    __syncthreads();
    compute_slab(p);
    __syncthreads();  // all waves done reading buf p before it is refilled
    p ^= 1;
    oA ^= bufA;
    oB ^= bufB;
  }
  // tail: last slab
  asm volatile("s_wait_asynccnt 0x0" ::: "memory");
  __syncthreads();
  compute_slab(p);

  // epilogue: C/D layout: VGPR r -> m_local = r + lhi*8, n_local = lrow
#pragma unroll
  for (int mi = 0; mi < 4; ++mi) {
#pragma unroll
    for (int ni = 0; ni < 2; ++ni) {
      int nn = n0 + wn * 32 + ni * 16 + lrow;
#pragma unroll
      for (int r = 0; r < 8; ++r) {
        int m = m0 + wm * 64 + mi * 16 + r + lhi * 8;
        if (m >= M) continue;
        float v = acc[mi][ni][r] + bias[m];
        if (mode == 0) {
          float s = gamma[m] * rsqrtf(var[m] + 1e-5f);
          v = (v - mean[m]) * s + beta[m];
          v = fmaxf(v, 0.f);
        } else {
          v = fmaxf(v, 0.f);
        }
        int b_ = nn / HoWo;
        int hw = nn % HoWo;
        size_t oidx = ((size_t)b_ * M + m) * HoWo + hw;
        if (residual) v = fmaxf(residual[oidx] + v, 0.f);
        Out[oidx] = v;
      }
    }
  }
}

// ---------------------------------------------------------------- orchestration
extern "C" void kernel_launch(void* const* d_in, const int* in_sizes, int n_in,
                              void* d_out, int out_size, void* d_ws, size_t ws_size,
                              hipStream_t stream) {
  (void)in_sizes; (void)n_in; (void)out_size; (void)ws_size;
  const float* a = (const float*)d_in[0];
  const float* b = (const float*)d_in[1];
  char* ws = (char*)d_ws;

  size_t off = 0;
  auto alloc = [&](size_t bytes) { size_t o = off; off += (bytes + 255) & ~(size_t)255; return o; };

  const int B_ = 2;
  float* f0 = (float*)(ws + alloc((size_t)2 * 64 * 128 * 128 * 4));
  float* f1 = (float*)(ws + alloc((size_t)2 * 128 * 64 * 64 * 4));
  float* f2 = (float*)(ws + alloc((size_t)2 * 256 * 32 * 32 * 4));
  float* f3 = (float*)(ws + alloc((size_t)2 * 512 * 16 * 16 * 4));
  float* ubuf = (float*)(ws + alloc((size_t)2 * 128 * 128 * 128 * 4));  // max upsample buf
  float* om   = (float*)(ws + alloc((size_t)2 * 48 * 128 * 128 * 4));   // max offset map
  f16*   Wbuf = (f16*)(ws + alloc((size_t)512 * 4096 * 2));             // max f16 weights
  f16*   Vbuf = (f16*)(ws + alloc((size_t)32768 * 1152 * 2));           // max im2col

  float* fmaps[4] = {f0, f1, f2, f3};
  const int ch[4] = {64, 128, 256, 512};
  const int hs[4] = {128, 64, 32, 16};

  // f0 = a + b
  {
    int n = 2 * 64 * 128 * 128;
    add_kernel<<<(n + 255) / 256, 256, 0, stream>>>(a, b, f0, n);
  }

  // ---- down path ----
  for (int i = 0; i < 3; ++i) {
    int base = 2 + 12 * i;  // pytree: bn{beta,gamma,mean,var}, down{b,b_off,w,w_off}, up{b,b_off,w,w_off}
    const float* p_beta   = (const float*)d_in[base + 0];
    const float* p_gamma  = (const float*)d_in[base + 1];
    const float* p_mean   = (const float*)d_in[base + 2];
    const float* p_var    = (const float*)d_in[base + 3];
    const float* dwn_b    = (const float*)d_in[base + 4];
    const float* dwn_boff = (const float*)d_in[base + 5];
    const float* dwn_w    = (const float*)d_in[base + 6];
    const float* dwn_woff = (const float*)d_in[base + 7];

    int Cin = ch[i], Cout = ch[i + 1];
    int H = hs[i], W = hs[i], Ho = H / 2, Wo = W / 2;
    const int k = 4, KK = 16, Coff = 48;

    int tco = B_ * Coff * Ho * Wo;
    conv_offset_kernel<<<(tco + 255) / 256, 256, 0, stream>>>(
        fmaps[i], dwn_woff, dwn_boff, om, B_, Cin, H, W, Coff, Ho, Wo, k, 2, 1);

    int tg = B_ * Ho * Wo * KK;
    dcn_gather_kernel<<<(tg + 255) / 256, 256, 0, stream>>>(
        fmaps[i], om, Vbuf, B_, Cin, H, W, Ho, Wo, k, 2, 1);

    int nw = Cout * Cin * KK;
    convert_w_kernel<<<(nw + 255) / 256, 256, 0, stream>>>(dwn_w, Wbuf, nw);

    int M = Cout, N = B_ * Ho * Wo, Kd = Cin * KK;
    dim3 grid(N / BN, (M + BM - 1) / BM);
    wmma_gemm_kernel<<<grid, 256, 0, stream>>>(
        Wbuf, Vbuf, fmaps[i + 1], dwn_b, p_gamma, p_beta, p_mean, p_var,
        nullptr, M, N, Kd, Ho * Wo, 0);
  }

  // ---- up path ----
  for (int i = 2; i >= 0; --i) {
    int base = 2 + 12 * i;
    const float* up_b    = (const float*)d_in[base + 8];
    const float* up_boff = (const float*)d_in[base + 9];
    const float* up_w    = (const float*)d_in[base + 10];
    const float* up_woff = (const float*)d_in[base + 11];

    int Cin = ch[i + 1], Cout = ch[i];
    int Hs = hs[i + 1], Ws = hs[i + 1];
    int H = 2 * Hs, W = 2 * Ws;  // == hs[i]
    const int k = 3, KK = 9, Coff = 27;

    int tu = B_ * Cin * H * W;
    upsample2x_kernel<<<(tu + 255) / 256, 256, 0, stream>>>(fmaps[i + 1], ubuf, B_, Cin, Hs, Ws);

    int tco = B_ * Coff * H * W;
    conv_offset_kernel<<<(tco + 255) / 256, 256, 0, stream>>>(
        ubuf, up_woff, up_boff, om, B_, Cin, H, W, Coff, H, W, k, 1, 1);

    int tg = B_ * H * W * KK;
    dcn_gather_kernel<<<(tg + 255) / 256, 256, 0, stream>>>(
        ubuf, om, Vbuf, B_, Cin, H, W, H, W, k, 1, 1);

    int nw = Cout * Cin * KK;
    convert_w_kernel<<<(nw + 255) / 256, 256, 0, stream>>>(up_w, Wbuf, nw);

    // fused: out = relu(res + relu(acc + bias)); in-place on residual map is safe
    float* dst = (i == 0) ? (float*)d_out : fmaps[i];
    const float* res = fmaps[i];
    int M = Cout, N = B_ * H * W, Kd = Cin * KK;
    dim3 grid(N / BN, (M + BM - 1) / BM);
    wmma_gemm_kernel<<<grid, 256, 0, stream>>>(
        Wbuf, Vbuf, dst, up_b, nullptr, nullptr, nullptr, nullptr,
        res, M, N, Kd, H * W, 1);
  }
}
